// BinaryConv2d_44057774522912
// MI455X (gfx1250) — matrix operand order repaired and checked by reading, not compiled, not run on hardware
//
#include <hip/hip_runtime.h>

typedef __attribute__((ext_vector_type(16))) __bf16 v16bf;
typedef __attribute__((ext_vector_type(8)))  float  v8f;

#define C_IN   128
#define C_OUT  256
#define HW     56
#define SP     (HW*HW)        // 3136
#define KTOT   (C_IN*9)       // 1152 = GEMM K
#define TILE_OC 128
#define TILE_P  128
#define KSTEP   32
#define KPAD    40            // shorts per LDS row: 80B = 16B-aligned + bank-spread
#define NSTEPS  (KTOT/KSTEP)  // 36
#define WB_ELEMS (C_OUT*KTOT) // 294912 bf16

// ---- prep: binarize latent fp32 weights -> +/-1 bf16, k-major = (kh*3+kw)*128+ic ----
__device__ __forceinline__ unsigned pack2w(float w0, float w1) {
    unsigned lo = (w0 >= 0.0f) ? 0x3F80u : 0xBF80u;
    unsigned hi = (w1 >= 0.0f) ? 0x3F80u : 0xBF80u;
    return lo | (hi << 16);
}
__device__ __forceinline__ unsigned pack2x(float v0, float v1) {
    unsigned short l = __builtin_bit_cast(unsigned short, (__bf16)v0);
    unsigned short h = __builtin_bit_cast(unsigned short, (__bf16)v1);
    return (unsigned)l | ((unsigned)h << 16);
}

__global__ __launch_bounds__(256)
void binarize_w(const float* __restrict__ w, unsigned short* __restrict__ wb) {
    const int idx  = (blockIdx.x * 256 + threadIdx.x) * 8;   // 144 blocks cover 294912
    const int oc   = idx / KTOT;
    const int r    = idx % KTOT;
    const int spix = r >> 7;          // kh*3+kw
    const int ic   = r & 127;
    const float* wp = w + (size_t)oc * KTOT + (size_t)ic * 9 + spix;
    uint4 q;
    q.x = pack2w(wp[0 * 9], wp[1 * 9]);
    q.y = pack2w(wp[2 * 9], wp[3 * 9]);
    q.z = pack2w(wp[4 * 9], wp[5 * 9]);
    q.w = pack2w(wp[6 * 9], wp[7 * 9]);
    *(uint4*)(wb + idx) = q;
}

struct PreT { uint4 a0, a1, b0, b1; };

template<bool USE_WB>
__device__ __forceinline__ PreT prefetch_step(int step,
        const float* __restrict__ x, const float* __restrict__ w,
        const unsigned short* __restrict__ wb,
        int oc_base, int lrow, int lkh, int nB, int ohB, int owB)
{
    PreT t;
    const int spix = step >> 2;
    const int c0   = (step & 3) * KSTEP;
    const int kh   = spix / 3, kw = spix - kh * 3;

    if (USE_WB) {
        const uint4* ap = (const uint4*)(wb + (size_t)(oc_base + lrow) * KTOT
                                            + spix * C_IN + c0 + lkh * 16);
        t.a0 = ap[0]; t.a1 = ap[1];
    } else {
        const float* wp = w + (size_t)(oc_base + lrow) * KTOT
                            + (size_t)(c0 + lkh * 16) * 9 + spix;
        t.a0.x = pack2w(wp[0*9],  wp[1*9]);   t.a0.y = pack2w(wp[2*9],  wp[3*9]);
        t.a0.z = pack2w(wp[4*9],  wp[5*9]);   t.a0.w = pack2w(wp[6*9],  wp[7*9]);
        t.a1.x = pack2w(wp[8*9],  wp[9*9]);   t.a1.y = pack2w(wp[10*9], wp[11*9]);
        t.a1.z = pack2w(wp[12*9], wp[13*9]);  t.a1.w = pack2w(wp[14*9], wp[15*9]);
    }

    const int ih = ohB - 1 + kh, iw = owB - 1 + kw;
    const bool valid = (ih >= 0) && (ih < HW) && (iw >= 0) && (iw < HW);
    const float* xp = x + (((size_t)nB * C_IN + c0 + lkh * 16) * HW + ih) * HW + iw;
#define XL(i) (valid ? xp[(size_t)(i) * SP] : 0.0f)
    t.b0.x = pack2x(XL(0),  XL(1));   t.b0.y = pack2x(XL(2),  XL(3));
    t.b0.z = pack2x(XL(4),  XL(5));   t.b0.w = pack2x(XL(6),  XL(7));
    t.b1.x = pack2x(XL(8),  XL(9));   t.b1.y = pack2x(XL(10), XL(11));
    t.b1.z = pack2x(XL(12), XL(13));  t.b1.w = pack2x(XL(14), XL(15));
#undef XL
    return t;
}

union FragU { v16bf v; struct { uint4 q0, q1; } s; };

template<bool USE_WB>
__global__ __launch_bounds__(256)
void bconv_wmma(const float* __restrict__ x, const float* __restrict__ w,
                const unsigned short* __restrict__ wb,
                const float* __restrict__ bias, float* __restrict__ out)
{
    __shared__ unsigned short As[2][TILE_OC][KPAD];  // 20 KB
    __shared__ unsigned short Bs[2][TILE_P][KPAD];   // 20 KB

    const int tid  = threadIdx.x;
    const int lane = tid & 31;
    const int wv   = tid >> 5;
    const int rg   = wv >> 1;        // oc group of 32 (0..3)
    const int cg   = wv & 1;         // pos group of 64 (0..1)

    const int p0      = blockIdx.x * TILE_P;
    const int oc_base = blockIdx.y * TILE_OC;

    const int lrow = tid & 127;      // A: oc row / B: position
    const int lkh  = tid >> 7;       // k-half of 16

    const int pB  = p0 + lrow;
    const int nB  = pB / SP;
    const int sB  = pB % SP;
    const int ohB = sB / HW;
    const int owB = sB % HW;

    const int hk = lane >> 4;
    const int mr = lane & 15;

    v8f acc00 = {}, acc01 = {}, acc02 = {}, acc03 = {};
    v8f acc10 = {}, acc11 = {}, acc12 = {}, acc13 = {};

    PreT t = prefetch_step<USE_WB>(0, x, w, wb, oc_base, lrow, lkh, nB, ohB, owB);
    {
        unsigned short* aDst = &As[0][lrow][lkh * 16];
        unsigned short* bDst = &Bs[0][lrow][lkh * 16];
        *(uint4*)aDst = t.a0; *(uint4*)(aDst + 8) = t.a1;
        *(uint4*)bDst = t.b0; *(uint4*)(bDst + 8) = t.b1;
    }
    __syncthreads();

    for (int step = 0; step < NSTEPS; ++step) {
        const int cur = step & 1;
        const bool more = (step + 1 < NSTEPS);
        if (more)
            t = prefetch_step<USE_WB>(step + 1, x, w, wb, oc_base, lrow, lkh, nB, ohB, owB);

        // ---- fragments: contiguous 16B chunks -> ds_load_b128 ----
        FragU af0, af1, bf0, bf1, bf2, bf3;
        {
            const unsigned short* ra0 = As[cur][rg * 32 + mr];
            const unsigned short* ra1 = As[cur][rg * 32 + 16 + mr];
            af0.s.q0 = *(const uint4*)&ra0[hk * 16];
            af0.s.q1 = *(const uint4*)&ra0[hk * 16 + 8];
            af1.s.q0 = *(const uint4*)&ra1[hk * 16];
            af1.s.q1 = *(const uint4*)&ra1[hk * 16 + 8];

            const unsigned short* rb0 = Bs[cur][cg * 64 + mr];
            const unsigned short* rb1 = Bs[cur][cg * 64 + 16 + mr];
            const unsigned short* rb2 = Bs[cur][cg * 64 + 32 + mr];
            const unsigned short* rb3 = Bs[cur][cg * 64 + 48 + mr];
            bf0.s.q0 = *(const uint4*)&rb0[hk * 16];
            bf0.s.q1 = *(const uint4*)&rb0[hk * 16 + 8];
            bf1.s.q0 = *(const uint4*)&rb1[hk * 16];
            bf1.s.q1 = *(const uint4*)&rb1[hk * 16 + 8];
            bf2.s.q0 = *(const uint4*)&rb2[hk * 16];
            bf2.s.q1 = *(const uint4*)&rb2[hk * 16 + 8];
            bf3.s.q0 = *(const uint4*)&rb3[hk * 16];
            bf3.s.q1 = *(const uint4*)&rb3[hk * 16 + 8];
        }

#define WMMA(acc, a, b) acc = __builtin_amdgcn_wmma_f32_16x16x32_bf16( \
            false, (a).v, false, (b).v, (short)0, acc, false, false)
        WMMA(acc00, af0, bf0); WMMA(acc01, af0, bf1);
        WMMA(acc02, af0, bf2); WMMA(acc03, af0, bf3);
        WMMA(acc10, af1, bf0); WMMA(acc11, af1, bf1);
        WMMA(acc12, af1, bf2); WMMA(acc13, af1, bf3);
#undef WMMA

        if (more) {
            const int nxt = cur ^ 1;
            unsigned short* aDst = &As[nxt][lrow][lkh * 16];
            unsigned short* bDst = &Bs[nxt][lrow][lkh * 16];
            *(uint4*)aDst = t.a0; *(uint4*)(aDst + 8) = t.a1;
            *(uint4*)bDst = t.b0; *(uint4*)(bDst + 8) = t.b1;
        }
        __syncthreads();
    }

    // ---- epilogue: C/D layout (lane half -> M+8), vector bias add, f32 stores ----
    const int mh   = hk * 8;
    const int ocr0 = oc_base + rg * 32 + mh;        // i=0 row base (32B aligned)
    const int ocr1 = ocr0 + 16;                     // i=1 row base
    const v8f bv0 = *(const v8f*)&bias[ocr0];
    const v8f bv1 = *(const v8f*)&bias[ocr1];
    acc00 += bv0; acc01 += bv0; acc02 += bv0; acc03 += bv0;
    acc10 += bv1; acc11 += bv1; acc12 += bv1; acc13 += bv1;

#define EPI(j, A0, A1) {                                              \
        const int p = p0 + cg * 64 + (j) * 16 + mr;                   \
        const int n = p / SP, s = p % SP;                             \
        float* op = out + (size_t)n * C_OUT * SP + s;                 \
        _Pragma("unroll")                                             \
        for (int q = 0; q < 8; ++q) {                                 \
            op[(size_t)(ocr0 + q) * SP] = A0[q];                      \
            op[(size_t)(ocr1 + q) * SP] = A1[q];                      \
        }                                                             \
    }
    EPI(0, acc00, acc10); EPI(1, acc01, acc11);
    EPI(2, acc02, acc12); EPI(3, acc03, acc13);
#undef EPI
}

extern "C" void kernel_launch(void* const* d_in, const int* in_sizes, int n_in,
                              void* d_out, int out_size, void* d_ws, size_t ws_size,
                              hipStream_t stream) {
    (void)in_sizes; (void)n_in; (void)out_size;
    const float* x    = (const float*)d_in[0];
    const float* wgt  = (const float*)d_in[1];
    const float* bias = (const float*)d_in[2];
    float* out = (float*)d_out;

    const int n_sp_tiles = (32 * SP) / TILE_P;   // 100352/128 = 784
    const int n_oc_tiles = C_OUT / TILE_OC;      // 2
    dim3 grid(n_sp_tiles, n_oc_tiles, 1);
    dim3 block(256, 1, 1);

    const size_t wb_bytes = (size_t)WB_ELEMS * sizeof(unsigned short); // 576 KB
    if (d_ws && ws_size >= wb_bytes) {
        unsigned short* wb = (unsigned short*)d_ws;
        binarize_w<<<dim3(WB_ELEMS / (256 * 8)), block, 0, stream>>>(wgt, wb);
        bconv_wmma<true><<<grid, block, 0, stream>>>(x, wgt, wb, bias, out);
    } else {
        bconv_wmma<false><<<grid, block, 0, stream>>>(x, wgt, nullptr, bias, out);
    }
}
